// SequenceMaskingAttention_88811333747326
// MI455X (gfx1250) — matrix operand order repaired and checked by reading, not compiled, not run on hardware
//
#include <hip/hip_runtime.h>
#include <hip/hip_bf16.h>

// Problem constants (B, N, C, H, D) = (2, 2048, 1024, 16, 64)
#define PB_B 2
#define PB_N 2048
#define PB_C 1024
#define PB_H 16
#define PB_D 64
#define QKV_COLS (3 * PB_H * PB_D)   // 3072
#define HD (PB_H * PB_D)             // 1024
#define MROWS (PB_B * PB_N)          // 4096
#define ATT_SCALE 0.125f             // D^-0.5

typedef __attribute__((ext_vector_type(8)))  __bf16 bf16x8;
typedef __attribute__((ext_vector_type(16))) __bf16 bf16x16;
typedef __attribute__((ext_vector_type(8)))  float  f32x8;

__device__ __forceinline__ unsigned short f2bf(float f) {
  union { float f; unsigned u; } v; v.f = f;
  unsigned r = v.u + 0x7FFFu + ((v.u >> 16) & 1u);  // RNE
  return (unsigned short)(r >> 16);
}

__device__ __forceinline__ f32x8 zero8() {
  f32x8 z = {0.f, 0.f, 0.f, 0.f, 0.f, 0.f, 0.f, 0.f};
  return z;
}

// 16-element bf16 fragment from two 16-byte chunks (two b128 loads).
__device__ __forceinline__ bf16x16 frag2(const unsigned short* lo,
                                         const unsigned short* hi) {
  bf16x8 a = *reinterpret_cast<const bf16x8*>(lo);
  bf16x8 b = *reinterpret_cast<const bf16x8*>(hi);
  bf16x16 r;
#pragma unroll
  for (int i = 0; i < 8; ++i) { r[i] = a[i]; r[i + 8] = b[i]; }
  return r;
}

__device__ __forceinline__ f32x8 wmma_bf16(bf16x16 a, bf16x16 b, f32x8 c) {
  return __builtin_amdgcn_wmma_f32_16x16x32_bf16(
      false, a, false, b, (short)0, c, false, false);
}

// A-fragment: row-major activation row, contraction contiguous.
__device__ __forceinline__ bf16x16 ldA(const unsigned short* arow, int kk,
                                       int koffA) {
  return frag2(arow + kk + 2 * koffA, arow + kk + 8 + 2 * koffA);
}
// B-fragment: n-major weight row, contraction contiguous (32B per lane).
__device__ __forceinline__ bf16x16 ldB(const unsigned short* bbase, int kk) {
  return frag2(bbase + kk, bbase + kk + 8);
}

// ---------------- conversion kernels ----------------
__global__ void cvt_f32_to_bf16_v4(const float* __restrict__ in,
                                   unsigned short* __restrict__ out, int n) {
  int n4 = n >> 2;
  int i = blockIdx.x * blockDim.x + threadIdx.x;
  int stride = gridDim.x * blockDim.x;
  for (; i < n4; i += stride) {
    float4 v = reinterpret_cast<const float4*>(in)[i];
    ushort4 o;
    o.x = f2bf(v.x); o.y = f2bf(v.y); o.z = f2bf(v.z); o.w = f2bf(v.w);
    reinterpret_cast<ushort4*>(out)[i] = o;
  }
}

// LDS-tiled transpose: in [rows][cols] f32 -> out [cols][rows] bf16.
// Coalesced reads and writes. grid = (cols/32, rows/32), block = 256.
__global__ __launch_bounds__(256)
void transpose_to_bf16(const float* __restrict__ in,
                       unsigned short* __restrict__ out, int rows, int cols) {
  __shared__ float tile[32][33];
  int tx = threadIdx.x & 31;
  int ty = threadIdx.x >> 5;
  int col = blockIdx.x * 32 + tx;
#pragma unroll
  for (int j = 0; j < 32; j += 8) {
    int row = blockIdx.y * 32 + ty + j;
    tile[ty + j][tx] = in[(size_t)row * cols + col];
  }
  __syncthreads();
  int ocol = blockIdx.y * 32 + tx;   // output col = source row
#pragma unroll
  for (int j = 0; j < 32; j += 8) {
    int orow = blockIdx.x * 32 + ty + j;  // output row = source col
    out[(size_t)orow * rows + ocol] = f2bf(tile[tx][ty + j]);
  }
}

// ---------------- GEMM1: x @ Wqkv, scatter to Q/K/V^T ----------------
__global__ __launch_bounds__(128)
void gemm_qkv(const unsigned short* __restrict__ X,      // [4096][1024] bf16
              const unsigned short* __restrict__ WT,     // [3072][1024] bf16
              unsigned short* __restrict__ Q,            // [B,H,N,D]
              unsigned short* __restrict__ K,            // [B,H,N,D]
              unsigned short* __restrict__ VT) {         // [B,H,D,N]
  const int ntiles = QKV_COLS / 64;                      // 48
  int wave = (blockIdx.x * blockDim.x + threadIdx.x) >> 5;
  int lane = threadIdx.x & 31;
  int mt = wave / ntiles, nt = wave % ntiles;
  if (mt >= MROWS / 16) return;
  int m0 = mt * 16, n0 = nt * 64;
  int mrow = lane & 15;
  int hf = (lane >> 4) & 1;
  int koffA = hf * 8;

  const unsigned short* arow = X + (size_t)(m0 + mrow) * PB_C;
  const unsigned short* bb = WT + (size_t)(n0 + (lane & 15)) * PB_C + hf * 16;
  const unsigned short* bb0 = bb;
  const unsigned short* bb1 = bb + (size_t)16 * PB_C;
  const unsigned short* bb2 = bb + (size_t)32 * PB_C;
  const unsigned short* bb3 = bb + (size_t)48 * PB_C;

  f32x8 acc[4];
#pragma unroll
  for (int t = 0; t < 4; ++t) acc[t] = zero8();

  // software-pipelined (double-buffered) k-loop
  bf16x16 aC = ldA(arow, 0, koffA);
  bf16x16 bC0 = ldB(bb0, 0), bC1 = ldB(bb1, 0);
  bf16x16 bC2 = ldB(bb2, 0), bC3 = ldB(bb3, 0);
  for (int kk = 32; kk < PB_C; kk += 32) {
    bf16x16 aN = ldA(arow, kk, koffA);
    bf16x16 bN0 = ldB(bb0, kk), bN1 = ldB(bb1, kk);
    bf16x16 bN2 = ldB(bb2, kk), bN3 = ldB(bb3, kk);
    acc[0] = wmma_bf16(aC, bC0, acc[0]);
    acc[1] = wmma_bf16(aC, bC1, acc[1]);
    acc[2] = wmma_bf16(aC, bC2, acc[2]);
    acc[3] = wmma_bf16(aC, bC3, acc[3]);
    aC = aN; bC0 = bN0; bC1 = bN1; bC2 = bN2; bC3 = bN3;
  }
  acc[0] = wmma_bf16(aC, bC0, acc[0]);
  acc[1] = wmma_bf16(aC, bC1, acc[1]);
  acc[2] = wmma_bf16(aC, bC2, acc[2]);
  acc[3] = wmma_bf16(aC, bC3, acc[3]);

  // epilogue: j = h*192 + d*3 + s   (reshape b n (h d 3))
#pragma unroll
  for (int t = 0; t < 4; ++t) {
#pragma unroll
    for (int r = 0; r < 8; ++r) {
      int row = m0 + r + 8 * hf;
      int bb_ = row / PB_N, nseq = row % PB_N;
      int j = n0 + t * 16 + (lane & 15);
      int h = j / (PB_D * 3);
      int rem = j % (PB_D * 3);
      int d = rem / 3, s = rem % 3;
      unsigned short v = f2bf(acc[t][r]);
      if (s == 0)      Q[((size_t)(bb_ * PB_H + h) * PB_N + nseq) * PB_D + d] = v;
      else if (s == 1) K[((size_t)(bb_ * PB_H + h) * PB_N + nseq) * PB_D + d] = v;
      else             VT[((size_t)(bb_ * PB_H + h) * PB_D + d) * PB_N + nseq] = v;
    }
  }
}

// ---------------- flash attention: one wave = 16 query rows ----------------
__global__ __launch_bounds__(32)
void attn_flash(const unsigned short* __restrict__ Q,
                const unsigned short* __restrict__ Kb,
                const unsigned short* __restrict__ VT,
                const float* __restrict__ POSB,          // [H,N,N] f32
                const unsigned char* __restrict__ MASK,  // [B,N] bool
                unsigned short* __restrict__ AO) {       // [B,N,H*D] bf16
  __shared__ alignas(16) unsigned short sP[16 * 32];     // P tile staging

  // batch fastest so the two batches share pos_bias tiles in L2
  int idx = blockIdx.x;
  int b = idx % PB_B;
  int t2 = idx / PB_B;
  int qblk = t2 % (PB_N / 16);
  int h = t2 / (PB_N / 16);
  int q0 = qblk * 16;

  int lane = threadIdx.x;
  int mrow = lane & 15;
  int hf = (lane >> 4) & 1;
  int koffA = hf * 16;
  int n = lane & 15;

  // persistent Q A-fragments (16x64 = two 16x32 fragments)
  const unsigned short* qrow =
      Q + ((size_t)(b * PB_H + h) * PB_N + q0 + mrow) * PB_D;
  bf16x16 aq0 = frag2(qrow + koffA, qrow + 8 + koffA);
  bf16x16 aq1 = frag2(qrow + 32 + koffA, qrow + 40 + koffA);

  const unsigned short* kbase = Kb + (size_t)(b * PB_H + h) * PB_N * PB_D;
  const unsigned short* vbase = VT + (size_t)(b * PB_H + h) * PB_D * PB_N;
  const float* pbh = POSB + (size_t)h * PB_N * PB_N;
  const unsigned char* mrowp = MASK + (size_t)b * PB_N;

  f32x8 o[4];
#pragma unroll
  for (int dt = 0; dt < 4; ++dt) o[dt] = zero8();
  float mrun[8], lrun[8];
#pragma unroll
  for (int r = 0; r < 8; ++r) { mrun[r] = -1e30f; lrun[r] = 0.f; }

  for (int k0 = 0; k0 < PB_N; k0 += 32) {
    // ---- issue ALL loads for this iteration up front ----
    const unsigned short* kp0 = kbase + (size_t)(k0 + n) * PB_D + hf * 16;
    const unsigned short* kp1 = kbase + (size_t)(k0 + 16 + n) * PB_D + hf * 16;
    bf16x16 kb00 = frag2(kp0, kp0 + 8);        // keys 0..15, d 0..31
    bf16x16 kb01 = frag2(kp0 + 32, kp0 + 40);  // keys 0..15, d 32..63
    bf16x16 kb10 = frag2(kp1, kp1 + 8);
    bf16x16 kb11 = frag2(kp1 + 32, kp1 + 40);

    float bv0[8], bv1[8];
#pragma unroll
    for (int r = 0; r < 8; ++r) {
      const float* pr = pbh + (size_t)(q0 + r + 8 * hf) * PB_N + k0;
      bv0[r] = pr[n];
      bv1[r] = pr[16 + n];
    }
    bool msk0 = mrowp[k0 + n] != 0;
    bool msk1 = mrowp[k0 + 16 + n] != 0;

    bf16x16 vb[4];
#pragma unroll
    for (int dt = 0; dt < 4; ++dt) {
      const unsigned short* vp =
          vbase + (size_t)(dt * 16 + n) * PB_N + k0 + hf * 16;
      vb[dt] = frag2(vp, vp + 8);
    }
    if (k0 + 32 < PB_N)
      __builtin_prefetch(pbh + (size_t)(q0 + 8 * hf) * PB_N + k0 + 32, 0, 0);

    // ---- S = Q K^T ----
    f32x8 s0 = zero8(), s1 = zero8();
    s0 = wmma_bf16(aq0, kb00, s0);
    s0 = wmma_bf16(aq1, kb01, s0);
    s1 = wmma_bf16(aq0, kb10, s1);
    s1 = wmma_bf16(aq1, kb11, s1);

    // ---- scale + bias + mask, online softmax ----
    float sv0[8], sv1[8];
#pragma unroll
    for (int r = 0; r < 8; ++r) {
      sv0[r] = msk0 ? -1e30f : (s0[r] * ATT_SCALE + bv0[r]);
      sv1[r] = msk1 ? -1e30f : (s1[r] * ATT_SCALE + bv1[r]);
    }

#pragma unroll
    for (int r = 0; r < 8; ++r) {
      float mx = fmaxf(sv0[r], sv1[r]);
#pragma unroll
      for (int off = 1; off < 16; off <<= 1)
        mx = fmaxf(mx, __shfl_xor(mx, off, 32));
      float mnew = fmaxf(mrun[r], mx);
      float alpha = __expf(mrun[r] - mnew);
      mrun[r] = mnew;
      float p0 = __expf(sv0[r] - mnew);
      float p1 = __expf(sv1[r] - mnew);
      float ps = p0 + p1;
#pragma unroll
      for (int off = 1; off < 16; off <<= 1) ps += __shfl_xor(ps, off, 32);
      lrun[r] = lrun[r] * alpha + ps;
#pragma unroll
      for (int dt = 0; dt < 4; ++dt) o[dt][r] *= alpha;
      // C-layout -> row-major 16x32 bf16 in LDS for A-fragment reload
      sP[(r + 8 * hf) * 32 + n] = f2bf(p0);
      sP[(r + 8 * hf) * 32 + 16 + n] = f2bf(p1);
    }

    // reload P in A-fragment layout (wave-internal DS ordering, no barrier)
    const unsigned short* prow = sP + mrow * 32 + koffA;
    bf16x16 pa = frag2(prow, prow + 8);

    // ---- O += P V (V fragments preloaded above) ----
    o[0] = wmma_bf16(pa, vb[0], o[0]);
    o[1] = wmma_bf16(pa, vb[1], o[1]);
    o[2] = wmma_bf16(pa, vb[2], o[2]);
    o[3] = wmma_bf16(pa, vb[3], o[3]);
  }

  // ---- normalize + store [B,N,H*D] bf16 ----
#pragma unroll
  for (int r = 0; r < 8; ++r) {
    float inv = 1.0f / lrun[r];
    int row = q0 + r + 8 * hf;
#pragma unroll
    for (int dt = 0; dt < 4; ++dt) {
      int col = h * PB_D + dt * 16 + n;
      AO[(size_t)(b * PB_N + row) * HD + col] = f2bf(o[dt][r] * inv);
    }
  }
}

// ---------------- GEMM2: attn_out @ Wout -> f32 output ----------------
__global__ __launch_bounds__(128)
void gemm_out(const unsigned short* __restrict__ A,     // [4096][1024] bf16
              const unsigned short* __restrict__ WT,    // [1024][1024] bf16 (n-major)
              float* __restrict__ OUT) {                // [4096][1024] f32
  const int ntiles = PB_C / 64;                         // 16
  int wave = (blockIdx.x * blockDim.x + threadIdx.x) >> 5;
  int lane = threadIdx.x & 31;
  int mt = wave / ntiles, nt = wave % ntiles;
  if (mt >= MROWS / 16) return;
  int m0 = mt * 16, n0 = nt * 64;
  int mrow = lane & 15;
  int hf = (lane >> 4) & 1;
  int koffA = hf * 8;

  const unsigned short* arow = A + (size_t)(m0 + mrow) * HD;
  const unsigned short* bb = WT + (size_t)(n0 + (lane & 15)) * HD + hf * 16;
  const unsigned short* bb0 = bb;
  const unsigned short* bb1 = bb + (size_t)16 * HD;
  const unsigned short* bb2 = bb + (size_t)32 * HD;
  const unsigned short* bb3 = bb + (size_t)48 * HD;

  f32x8 acc[4];
#pragma unroll
  for (int t = 0; t < 4; ++t) acc[t] = zero8();

  bf16x16 aC = ldA(arow, 0, koffA);
  bf16x16 bC0 = ldB(bb0, 0), bC1 = ldB(bb1, 0);
  bf16x16 bC2 = ldB(bb2, 0), bC3 = ldB(bb3, 0);
  for (int kk = 32; kk < HD; kk += 32) {
    bf16x16 aN = ldA(arow, kk, koffA);
    bf16x16 bN0 = ldB(bb0, kk), bN1 = ldB(bb1, kk);
    bf16x16 bN2 = ldB(bb2, kk), bN3 = ldB(bb3, kk);
    acc[0] = wmma_bf16(aC, bC0, acc[0]);
    acc[1] = wmma_bf16(aC, bC1, acc[1]);
    acc[2] = wmma_bf16(aC, bC2, acc[2]);
    acc[3] = wmma_bf16(aC, bC3, acc[3]);
    aC = aN; bC0 = bN0; bC1 = bN1; bC2 = bN2; bC3 = bN3;
  }
  acc[0] = wmma_bf16(aC, bC0, acc[0]);
  acc[1] = wmma_bf16(aC, bC1, acc[1]);
  acc[2] = wmma_bf16(aC, bC2, acc[2]);
  acc[3] = wmma_bf16(aC, bC3, acc[3]);

#pragma unroll
  for (int t = 0; t < 4; ++t) {
#pragma unroll
    for (int r = 0; r < 8; ++r) {
      int row = m0 + r + 8 * hf;
      OUT[(size_t)row * PB_C + n0 + t * 16 + (lane & 15)] = acc[t][r];
    }
  }
}

// ---------------- host launcher ----------------
extern "C" void kernel_launch(void* const* d_in, const int* in_sizes, int n_in,
                              void* d_out, int out_size, void* d_ws, size_t ws_size,
                              hipStream_t stream) {
  (void)in_sizes; (void)n_in; (void)out_size; (void)ws_size;
  const float* x    = (const float*)d_in[0];           // [B,N,C]
  const float* posb = (const float*)d_in[1];           // [H,N,N]
  const float* wqkv = (const float*)d_in[2];           // [C, 3HD]
  const float* wout = (const float*)d_in[3];           // [HD, C]
  const unsigned char* mask = (const unsigned char*)d_in[4]; // [B,N]
  float* out = (float*)d_out;                          // [B,N,C]

  char* ws = (char*)d_ws;
  size_t off = 0;
  auto take = [&](size_t bytes) -> void* {
    void* p = ws + off;
    off += (bytes + 255) & ~(size_t)255;
    return p;
  };
  unsigned short* xb    = (unsigned short*)take((size_t)MROWS * PB_C * 2);
  unsigned short* wqkvT = (unsigned short*)take((size_t)QKV_COLS * PB_C * 2);
  unsigned short* woutT = (unsigned short*)take((size_t)PB_C * HD * 2);
  unsigned short* Qb    = (unsigned short*)take((size_t)MROWS * HD * 2);
  unsigned short* Kb    = (unsigned short*)take((size_t)MROWS * HD * 2);
  unsigned short* VTb   = (unsigned short*)take((size_t)MROWS * HD * 2);
  unsigned short* AOb   = (unsigned short*)take((size_t)MROWS * HD * 2);

  cvt_f32_to_bf16_v4<<<4096, 256, 0, stream>>>(x, xb, MROWS * PB_C);
  transpose_to_bf16<<<dim3(QKV_COLS / 32, PB_C / 32), 256, 0, stream>>>(
      wqkv, wqkvT, PB_C, QKV_COLS);
  transpose_to_bf16<<<dim3(PB_C / 32, HD / 32), 256, 0, stream>>>(
      wout, woutT, HD, PB_C);

  // 12288 output tiles (16x64), 4 waves/block
  gemm_qkv<<<3072, 128, 0, stream>>>(xb, wqkvT, Qb, Kb, VTb);

  // B*H*(N/16) = 4096 wave-blocks
  attn_flash<<<PB_B * PB_H * (PB_N / 16), 32, 0, stream>>>(Qb, Kb, VTb, posb,
                                                           mask, AOb);

  // 4096 output tiles (16x64), 4 waves/block
  gemm_out<<<1024, 128, 0, stream>>>(AOb, woutT, out);
}
